// ARC_87058987090488
// MI455X (gfx1250) — matrix operand order repaired and checked by reading, not compiled, not run on hardware
//
#include <hip/hip_runtime.h>
#include <hip/hip_bf16.h>

// ---------------- CDNA5 WMMA types ----------------
typedef __attribute__((ext_vector_type(16))) __bf16       v16bf;
typedef __attribute__((ext_vector_type(8)))  float        v8f;
typedef __attribute__((ext_vector_type(8)))  unsigned int v8u;

// ---------------- problem constants ----------------
constexpr int HID = 128, GH = 8, GW = 8, IMGH = 64, IMGW = 64, STEPS = 16;
constexpr int SPB = 16;          // samples per workgroup
constexpr int NTHREADS = 128;    // 4 waves of 32

// ---------------- LDS layout (bytes) ----------------
// strides in elements; padded so every fragment ds_load_b64 is 8B aligned
constexpr int WIH_ST = 68;   // W_ih rows: 64 data + pad
constexpr int WHH_ST = 132;  // W_hh rows: 128 data + pad
constexpr int F_ST   = 64;   // filterbank rows (to-major [t][i])
constexpr int IMG_ST = 68;   // transposed image rows [w][h]
constexpr int G_ST   = 68;   // per-wave g scratch rows [t][w]
constexpr int X_ST   = 68;   // x rows [m][64]
constexpr int HB_ST  = 132;  // bf16 h rows [m][128]

constexpr int OFF_WIH  = 0;
constexpr int OFF_WHH  = OFF_WIH  + 512 * WIH_ST * 2;     // 69632
constexpr int OFF_BIAS = OFF_WHH  + 512 * WHH_ST * 2;     // 204800
constexpr int OFF_HB   = OFF_BIAS + 512 * 4;              // 206848
constexpr int OFF_C    = OFF_HB   + SPB * HB_ST * 2;      // 211072
constexpr int OFF_X    = OFF_C    + SPB * HID * 4;        // 219264
constexpr int OFF_GP   = OFF_X    + SPB * X_ST * 2;       // 221440
constexpr int OFF_FH   = OFF_GP   + 256;                  // 221696
constexpr int OFF_FW   = OFF_FH   + SPB * GH * F_ST * 2;  // 238080
constexpr int OFF_IMG  = OFF_FW   + SPB * GW * F_ST * 2;  // 254464
constexpr int OFF_GB   = OFF_IMG  + 4 * IMGW * IMG_ST * 2;// 289280
constexpr int OFF_GATE = OFF_GB   + 4 * GH * G_ST * 2;    // 293632
constexpr int SMEM_BYTES = OFF_GATE + 4 * SPB * HID * 4;  // 326400
static_assert(SMEM_BYTES <= 320 * 1024, "exceeds 320KB WGP LDS");

// ---------------- helpers ----------------
__device__ __forceinline__ unsigned short f2bf(float f) {
  unsigned u = __builtin_bit_cast(unsigned, f);
  unsigned r = u + 0x7FFFu + ((u >> 16) & 1u);   // round-to-nearest-even
  return (unsigned short)(r >> 16);
}
__device__ __forceinline__ float bf2f(unsigned short h) {
  unsigned u = ((unsigned)h) << 16;
  return __builtin_bit_cast(float, u);
}
__device__ __forceinline__ float sigm(float x) { return 1.0f / (1.0f + __expf(-x)); }

__device__ __forceinline__ v8f wmma_bf16(v16bf a, v16bf b, v8f c) {
  // D = A(16x32 bf16) * B(32x16 bf16) + C(16x16 f32)
  return __builtin_amdgcn_wmma_f32_16x16x32_bf16(false, a, false, b, (short)0, c,
                                                 false, false);
}

// A fragment (16x32 bf16). Source is LDS, row-major [row][k], K-pairs contiguous.
// A layout: lanes 0-15 row M=lane, lanes 16-31 same rows, K offset +8;
// VGPR groups hold K {0-3, 4-7, 16-19, 20-23} (+hi*8, +kc*32).
__device__ __forceinline__ v16bf lds_frag_a(const char* smem, int byteOff,
                                            int strideElems, int row, int kc, bool valid) {
  const int lane = threadIdx.x & 31;
  const int hi   = lane >> 4;
  v8u u = {0, 0, 0, 0, 0, 0, 0, 0};
  if (valid) {
    const char* p = smem + byteOff + (size_t)row * (size_t)(strideElems * 2)
                  + (size_t)((kc * 32 + hi * 8) * 2);
    uint2 g0 = *(const uint2*)(p + 0);
    uint2 g1 = *(const uint2*)(p + 8);
    uint2 g2 = *(const uint2*)(p + 32);
    uint2 g3 = *(const uint2*)(p + 40);
    u[0] = g0.x; u[1] = g0.y; u[2] = g1.x; u[3] = g1.y;
    u[4] = g2.x; u[5] = g2.y; u[6] = g3.x; u[7] = g3.y;
  }
  return __builtin_bit_cast(v16bf, u);
}

// B fragment (32x16 bf16). Source is LDS holding B-transposed: [n][k], K contiguous.
// B layout: lane column N=lane%16; VGPR groups hold K {0-3, 4-7, 16-19, 20-23} (+hi*8, +kc*32).
__device__ __forceinline__ v16bf lds_frag_b(const char* smem, int byteOff,
                                            int strideElems, int col, int kc, bool valid) {
  const int lane = threadIdx.x & 31;
  const int hi   = lane >> 4;
  v8u u = {0, 0, 0, 0, 0, 0, 0, 0};
  if (valid) {
    const char* p = smem + byteOff + (size_t)col * (size_t)(strideElems * 2)
                  + (size_t)((kc * 32 + hi * 8) * 2);
    uint2 g0 = *(const uint2*)(p + 0);
    uint2 g1 = *(const uint2*)(p + 8);
    uint2 g2 = *(const uint2*)(p + 32);
    uint2 g3 = *(const uint2*)(p + 40);
    u[0] = g0.x; u[1] = g0.y; u[2] = g1.x; u[3] = g1.y;
    u[4] = g2.x; u[5] = g2.y; u[6] = g3.x; u[7] = g3.y;
  }
  return __builtin_bit_cast(v16bf, u);
}

// ---------------- persistent ARC kernel: 16 samples per WG, all 16 steps ----------------
__global__ void __launch_bounds__(NTHREADS)
arc_forward_kernel(const float* __restrict__ img_pairs,
                   const float* __restrict__ W_ih, const float* __restrict__ W_hh,
                   const float* __restrict__ b_ih, const float* __restrict__ b_hh,
                   const float* __restrict__ W_g,  const float* __restrict__ b_g,
                   float* __restrict__ out) {
  extern __shared__ char smem[];
  const int tid  = threadIdx.x;
  const int lane = tid & 31;
  const int wave = tid >> 5;
  const int s0   = blockIdx.x * SPB;   // first global sample of this WG

  // ---- one-time staging: bf16 weights + combined bias + zero state ----
  for (int idx = tid; idx < 512 * 64; idx += NTHREADS) {
    int n = idx >> 6, k = idx & 63;
    *(unsigned short*)(smem + OFF_WIH + (n * WIH_ST + k) * 2) = f2bf(W_ih[idx]);
  }
  for (int idx = tid; idx < 512 * 128; idx += NTHREADS) {
    int n = idx >> 7, k = idx & 127;
    *(unsigned short*)(smem + OFF_WHH + (n * WHH_ST + k) * 2) = f2bf(W_hh[idx]);
  }
  for (int j = tid; j < 512; j += NTHREADS)
    *(float*)(smem + OFF_BIAS + j * 4) = b_ih[j] + b_hh[j];
  for (int idx = tid; idx < SPB * HB_ST; idx += NTHREADS)
    *(unsigned short*)(smem + OFF_HB + idx * 2) = 0;
  for (int idx = tid; idx < SPB * HID; idx += NTHREADS)
    *(float*)(smem + OFF_C + idx * 4) = 0.0f;
  __syncthreads();

  for (int step = 0; step < STEPS; ++step) {
    // ---- (1) gp = tanh(h @ W_g^T + b_g): 16 samples x 3 outputs ----
    if (tid < 48) {
      int bb = tid / 3, rg = tid % 3;
      float acc = b_g[rg];
      const unsigned short* hb = (const unsigned short*)(smem + OFF_HB) + bb * HB_ST;
      #pragma unroll 4
      for (int k = 0; k < HID; ++k) acc += bf2f(hb[k]) * W_g[rg * HID + k];
      *(float*)(smem + OFF_GP + (bb * 3 + rg) * 4) = tanhf(acc);
    }
    __syncthreads();

    // ---- (2) Cauchy filterbanks (stored to-major [b][t][i], bf16) ----
    for (int task = tid; task < 256; task += NTHREADS) {
      int bb = task >> 4, filt = (task >> 3) & 1, t = task & 7;
      const float* gpb = (const float*)(smem + OFF_GP) + bb * 3;
      float ctr = gpb[filt], ad = fabsf(gpb[2]);
      float centers = 31.5f * (ctr + 1.0f);
      float deltas  = 8.0f * (1.0f - ad);
      float invg    = __expf(2.0f * ad - 1.0f);          // 1/gamma
      float gpix    = centers + deltas * ((float)t - 3.5f);
      float s = 0.0f;
      #pragma unroll 8
      for (int i = 0; i < 64; ++i) {
        float u = ((float)i - gpix) * invg;
        s += invg * 0.318309886f / (1.0f + u * u);
      }
      float norm = 1.0f / (s + 1.0e-4f);
      unsigned short* Fp =
          (unsigned short*)(smem + (filt ? OFF_FW : OFF_FH)) + (bb * 8 + t) * F_ST;
      #pragma unroll 8
      for (int i = 0; i < 64; ++i) {
        float u = ((float)i - gpix) * invg;
        Fp[i] = f2bf(invg * 0.318309886f / (1.0f + u * u) * norm);
      }
    }
    __syncthreads();

    // ---- (3) glimpse: each wave does 4 samples; g = Fh^T*img, x = g*Fw ----
    {
      const int t = lane & 15;
      const bool tval = t < 8;
      char* imgT = smem + OFF_IMG + wave * (IMGW * IMG_ST * 2);
      for (int ss = 0; ss < 4; ++ss) {
        const int bb = wave * 4 + ss;
        const float4* iv =
            (const float4*)(img_pairs + (((size_t)(s0 + bb)) * 2 + (size_t)(step & 1)) * 4096);
        // stage image transposed as bf16: imgT[w][h]
        #pragma unroll 4
        for (int rep = 0; rep < 32; ++rep) {
          int idx = rep * 32 + lane;
          float4 v = iv[idx];
          int flat = idx * 4, hh = flat >> 6, w0 = flat & 63;
          unsigned short* ip = (unsigned short*)imgT;
          ip[(w0 + 0) * IMG_ST + hh] = f2bf(v.x);
          ip[(w0 + 1) * IMG_ST + hh] = f2bf(v.y);
          ip[(w0 + 2) * IMG_ST + hh] = f2bf(v.z);
          ip[(w0 + 3) * IMG_ST + hh] = f2bf(v.w);
        }
        asm volatile("s_wait_dscnt 0x0" ::: "memory");

        // g[t][w] = sum_h Fh[t][h] * img[h][w]   (M=8 padded to 16, N=64, K=64)
        unsigned short* gb = (unsigned short*)(smem + OFF_GB) + wave * (GH * G_ST);
        #pragma unroll
        for (int nt = 0; nt < 4; ++nt) {
          v8f acc = {0, 0, 0, 0, 0, 0, 0, 0};
          #pragma unroll
          for (int kc = 0; kc < 2; ++kc) {
            v16bf a = lds_frag_a(smem, OFF_FH + bb * 8 * F_ST * 2, F_ST, t, kc, tval);
            v16bf b = lds_frag_b(smem, OFF_IMG + wave * IMGW * IMG_ST * 2, IMG_ST,
                                 nt * 16 + (lane & 15), kc, true);
            acc = wmma_bf16(a, b, acc);
          }
          if (lane < 16) {   // D rows 0..7 live in lanes 0..15, VGPR r = row
            #pragma unroll
            for (int r = 0; r < 8; ++r) gb[r * G_ST + nt * 16 + lane] = f2bf(acc[r]);
          }
        }
        asm volatile("s_wait_dscnt 0x0" ::: "memory");

        // gg[t][s] = sum_w g[t][w] * Fw[s][w]^T  (M=8 pad, N=8 pad, K=64)
        v8f acc2 = {0, 0, 0, 0, 0, 0, 0, 0};
        #pragma unroll
        for (int kc = 0; kc < 2; ++kc) {
          v16bf a = lds_frag_a(smem, OFF_GB + wave * GH * G_ST * 2, G_ST, t, kc, tval);
          v16bf b = lds_frag_b(smem, OFF_FW + bb * 8 * F_ST * 2, F_ST, lane & 15, kc,
                               (lane & 15) < 8);
          acc2 = wmma_bf16(a, b, acc2);
        }
        if (lane < 8) {      // x[bb][t*8+s], t = VGPR r, s = lane
          unsigned short* xp = (unsigned short*)(smem + OFF_X) + bb * X_ST;
          #pragma unroll
          for (int r = 0; r < 8; ++r) xp[r * 8 + lane] = f2bf(acc2[r]);
        }
      }
    }
    __syncthreads();

    // ---- (4) gates = x@W_ih^T + h@W_hh^T + bias : wave w owns gate kind w ----
    {
      const int m  = lane & 15;
      const int hi = lane >> 4;
      v16bf ax0 = lds_frag_a(smem, OFF_X, X_ST, m, 0, true);
      v16bf ax1 = lds_frag_a(smem, OFF_X, X_ST, m, 1, true);
      v16bf ah0 = lds_frag_a(smem, OFF_HB, HB_ST, m, 0, true);
      v16bf ah1 = lds_frag_a(smem, OFF_HB, HB_ST, m, 1, true);
      v16bf ah2 = lds_frag_a(smem, OFF_HB, HB_ST, m, 2, true);
      v16bf ah3 = lds_frag_a(smem, OFF_HB, HB_ST, m, 3, true);
      float* gatesBank = (float*)(smem + OFF_GATE);
      #pragma unroll 2
      for (int nt = 0; nt < 8; ++nt) {
        const int ntAbs = wave * 8 + nt;       // absolute N-tile over 512 gates
        const int col   = ntAbs * 16 + m;
        v8f acc = {0, 0, 0, 0, 0, 0, 0, 0};
        acc = wmma_bf16(ax0, lds_frag_b(smem, OFF_WIH, WIH_ST, col, 0, true), acc);
        acc = wmma_bf16(ax1, lds_frag_b(smem, OFF_WIH, WIH_ST, col, 1, true), acc);
        acc = wmma_bf16(ah0, lds_frag_b(smem, OFF_WHH, WHH_ST, col, 0, true), acc);
        acc = wmma_bf16(ah1, lds_frag_b(smem, OFF_WHH, WHH_ST, col, 1, true), acc);
        acc = wmma_bf16(ah2, lds_frag_b(smem, OFF_WHH, WHH_ST, col, 2, true), acc);
        acc = wmma_bf16(ah3, lds_frag_b(smem, OFF_WHH, WHH_ST, col, 3, true), acc);
        float bias = *(const float*)(smem + OFF_BIAS + col * 4);
        #pragma unroll
        for (int r = 0; r < 8; ++r) {
          int mm = r + hi * 8;                 // sample row
          gatesBank[(wave * SPB + mm) * HID + nt * 16 + m] = acc[r] + bias;
        }
      }
    }
    __syncthreads();

    // ---- (5) pointwise LSTM update; thread tid owns hidden unit j=tid ----
    {
      const int j = tid;
      const float* gbk = (const float*)(smem + OFF_GATE);
      #pragma unroll 4
      for (int mm = 0; mm < SPB; ++mm) {
        float gi = gbk[(0 * SPB + mm) * HID + j];
        float gf = gbk[(1 * SPB + mm) * HID + j];
        float gg = gbk[(2 * SPB + mm) * HID + j];
        float go = gbk[(3 * SPB + mm) * HID + j];
        float* cp = (float*)(smem + OFF_C) + mm * HID + j;
        float c = sigm(gf) * (*cp) + sigm(gi) * tanhf(gg);
        float h = sigm(go) * tanhf(c);
        *cp = c;
        ((unsigned short*)(smem + OFF_HB))[mm * HB_ST + j] = f2bf(h);
        if (step == STEPS - 1) out[(size_t)(s0 + mm) * HID + j] = h;
      }
    }
    __syncthreads();
  }
}

extern "C" void kernel_launch(void* const* d_in, const int* in_sizes, int n_in,
                              void* d_out, int out_size, void* d_ws, size_t ws_size,
                              hipStream_t stream) {
  const float* img  = (const float*)d_in[0];
  const float* W_ih = (const float*)d_in[1];
  const float* W_hh = (const float*)d_in[2];
  const float* b_ih = (const float*)d_in[3];
  const float* b_hh = (const float*)d_in[4];
  const float* W_g  = (const float*)d_in[5];
  const float* b_g  = (const float*)d_in[6];
  float* out = (float*)d_out;

  const int nSamples = in_sizes[0] / (2 * IMGH * IMGW);   // 4096
  const int nWG = nSamples / SPB;                         // 256 WGs, one per WGP
  arc_forward_kernel<<<nWG, NTHREADS, SMEM_BYTES, stream>>>(img, W_ih, W_hh, b_ih,
                                                            b_hh, W_g, b_g, out);
}
